// Protein_graph_60876866454166
// MI455X (gfx1250) — matrix operand order repaired and checked by reading, not compiled, not run on hardware
//
#include <hip/hip_runtime.h>
#include <hip/hip_bf16.h>
#include <math.h>

typedef __attribute__((ext_vector_type(16))) _Float16 v16h;
typedef __attribute__((ext_vector_type(8)))  float    v8f;

#define PP    1512
#define NN    400
#define EPER  2000
#define DIN   7
#define HH    64
#define KSEL  200
#define MM    65536
#define NEDGE (PP*EPER)
#define BN_EPS 1e-5f

// ---- CDNA5 async global->LDS copy (per-lane, 16B), ASYNCcnt-tracked ----
__device__ __forceinline__ void async_copy_b128(unsigned lds_off, unsigned long long gaddr) {
    asm volatile("global_load_async_to_lds_b128 %0, %1, off"
                 :: "v"(lds_off), "v"(gaddr) : "memory");
}
__device__ __forceinline__ void wait_asynccnt0() {
    asm volatile("s_wait_asynccnt 0" ::: "memory");
}
__device__ __forceinline__ unsigned lds_off_of(const void* p) {
    return (unsigned)(unsigned long long)p;   // generic shared ptr: low 32 bits = LDS byte addr
}

// k-index maps for 16-bit WMMA operands (ISA 7.12.2, wave32)
__device__ __forceinline__ int amap(int e, int khalf) {
    return (e < 8) ? (8 * khalf + e) : (16 + 8 * khalf + (e - 8));
}
__device__ __forceinline__ int bmap(int e, int khalf) {
    return (e < 8) ? (8 * khalf + e) : (16 + 8 * khalf + (e - 8));
}

// ---------------- Kernel A: per-protein fused GNN ----------------
// dynamic LDS layout (bytes):
#define OFF_H3    0        // h3  [400][64] f32 = 102400 (post-BN activations; written phase 4)
#define OFF_ESTG  32768    // edge staging: src[2000] + dst[2000] int32 = 16000 (inside h3, dead by phase 4)
#define OFF_H16   102400   // h16 [400][64] f16 = 51200  (fc1 WMMA A operand)
#define OFF_XA    153600   // xa  [400][8]  f32 = 12800  | reused: fc1 w16 f16 [64][64]=8192
#define OFF_AGG   166400   // agg [400][8]  f32 = 12800  | reused: srel/sroot/scacc/wsel 4x1600
#define SMEM_A    179200

__global__ __launch_bounds__(256) void protein_gnn_kernel(
    const float* __restrict__ x, const int* __restrict__ ei,
    const float* __restrict__ sage_wl, const float* __restrict__ sage_wr,
    const float* __restrict__ sage_b,
    const float* __restrict__ fc1_w, const float* __restrict__ fc1_b,
    const float* __restrict__ bn_g, const float* __restrict__ bn_b,
    const float* __restrict__ bn_rm, const float* __restrict__ bn_rv,
    const float* __restrict__ sag_wrel, const float* __restrict__ sag_brel,
    const float* __restrict__ sag_wroot,
    float* __restrict__ emb_out)
{
    extern __shared__ char smem[];
    float*    h3    = (float*)(smem + OFF_H3);      // [400][64]
    int*      esrc  = (int*)(smem + OFF_ESTG);      // [2000]
    int*      edst  = (int*)(smem + OFF_ESTG + 8000);
    _Float16* h16   = (_Float16*)(smem + OFF_H16);  // [400][64]
    float*    xa    = (float*)(smem + OFF_XA);      // [400][8] (padded)
    _Float16* w16   = (_Float16*)(smem + OFF_XA);   // [64][64] after phase 2
    float*    agg   = (float*)(smem + OFF_AGG);     // [400][8]
    float*    srel  = (float*)(smem + OFF_AGG);             // [400]
    float*    sroot = (float*)(smem + OFF_AGG + 1600);      // [400]
    float*    scacc = (float*)(smem + OFF_AGG + 3200);      // [400]
    float*    wsel  = (float*)(smem + OFF_AGG + 4800);      // [400]

    __shared__ float wl[DIN * HH], wr[DIN * HH];
    __shared__ float sb[HH], fb[HH], bnsc[HH], bnbi[HH], wrelS[HH], wrootS[HH];
    __shared__ float scoreS[NN];
    __shared__ float part[4 * HH];
    __shared__ unsigned epack[EPER];   // packed u16 local (src,dst) for second segment-sum pass

    const int p = blockIdx.x;
    const int t = threadIdx.x;
    const int lane = t & 31;
    const int wave = __builtin_amdgcn_readfirstlane(t >> 5);  // wave-uniform -> SALU loop control
    const int mrow = lane & 15, khalf = lane >> 4;

    // ---- phase 0a: kick off async DMA of this protein's edge lists into LDS ----
    // (16B/lane chunks; overlapped with the VALU staging below; ASYNCcnt-tracked)
    const int ebase = p * EPER;
    {
        unsigned l0 = lds_off_of(esrc);
        unsigned l1 = lds_off_of(edst);
        unsigned long long g0 = (unsigned long long)(const void*)(ei + ebase);
        unsigned long long g1 = (unsigned long long)(const void*)(ei + NEDGE + ebase);
        for (int c = t; c < 500; c += 256) {            // 500 * 16B = 8000B per list
            async_copy_b128(l0 + 16u * c, g0 + 16ull * c);
            async_copy_b128(l1 + 16u * c, g1 + 16ull * c);
        }
    }

    // ---- phase 0b: stage x / weights while the DMA is in flight ----
    for (int i = t; i < NN * DIN; i += 256) {
        int node = i / DIN, k = i - node * DIN;
        xa[node * 8 + k] = x[(size_t)(p * NN + node) * DIN + k];
    }
    for (int i = t; i < NN * 8; i += 256) agg[i] = 0.0f;
    for (int i = t; i < DIN * HH; i += 256) { wl[i] = sage_wl[i]; wr[i] = sage_wr[i]; }
    if (t < HH) {
        float sc = bn_g[t] * rsqrtf(bn_rv[t] + BN_EPS);
        bnsc[t] = sc; bnbi[t] = bn_b[t] - bn_rm[t] * sc;
        sb[t] = sage_b[t]; fb[t] = fc1_b[t];
        wrelS[t] = sag_wrel[t]; wrootS[t] = sag_wroot[t];
    }
    wait_asynccnt0();           // each wave drains its own async copies
    __syncthreads();            // then LDS edge data is visible to all waves

    // ---- phase 1: SAGE sum-aggregation via LDS atomics; pack local edge ids ----
    const int pbase = p * NN;
    for (int e = t; e < EPER; e += 256) {
        int s = esrc[e] - pbase;
        int d = edst[e] - pbase;
        epack[e] = (unsigned)s | ((unsigned)d << 16);
#pragma unroll
        for (int k = 0; k < DIN; k++)
            atomicAdd(&agg[d * 8 + k], xa[s * 8 + k]);
    }
    __syncthreads();

    // ---- phase 2: h1 = agg@wl + x@wr + b  (K=7, VALU) -> f16 ----
    for (int i = t; i < NN * HH; i += 256) {
        int node = i >> 6, f = i & 63;
        float acc = sb[f];
#pragma unroll
        for (int k = 0; k < DIN; k++)
            acc += agg[node * 8 + k] * wl[k * HH + f] + xa[node * 8 + k] * wr[k * HH + f];
        h16[i] = (_Float16)acc;
    }
    __syncthreads();

    // ---- phase 3: stage fc1 weights as f16 (xa region is dead) ----
    for (int i = t; i < HH * HH; i += 256) w16[i] = (_Float16)fc1_w[i];
    __syncthreads();

    // ---- phase 4: fc1 GEMM [400,64]x[64,64] via v_wmma_f32_16x16x32_f16 ----
    // 25 row-tiles x 4 col-tiles = 100 tiles, 8 waves (scalar loop: EXEC stays all-ones)
    for (int tile = wave; tile < 100; tile += 8) {
        int mt = (tile >> 2) * 16;
        int nt = (tile & 3) * 16;
        v8f c = {};
#pragma unroll
        for (int kb = 0; kb < HH; kb += 32) {
            v16h a, b;
#pragma unroll
            for (int e = 0; e < 16; e++) {
                a[e] = h16[(mt + mrow) * HH + kb + amap(e, khalf)];
                b[e] = w16[(kb + bmap(e, khalf)) * HH + nt + mrow];
            }
            c = __builtin_amdgcn_wmma_f32_16x16x32_f16(false, a, false, b,
                                                       (short)0, c, false, false);
        }
        int f = nt + mrow;
        float sc = bnsc[f], bi = bnbi[f], bv = fb[f];
#pragma unroll
        for (int r = 0; r < 8; r++) {
            int m = mt + r + 8 * khalf;
            float v = fmaxf(c[r] + bv, 0.0f);      // ReLU
            h3[m * HH + f] = v * sc + bi;          // BatchNorm (eval)
        }
    }
    __syncthreads();

    // ---- phase 5: score dots (agg region dead -> reuse) ----
    for (int node = t; node < NN; node += 256) {
        float a1 = 0.0f, a2 = 0.0f;
#pragma unroll 8
        for (int f = 0; f < HH; f++) {
            float hv = h3[node * HH + f];
            a1 += hv * wrelS[f]; a2 += hv * wrootS[f];
        }
        srel[node] = a1; sroot[node] = a2; scacc[node] = 0.0f;
    }
    __syncthreads();

    // ---- phase 6: scalar segment-sum (sagg@wrel == scatter of srel), edges from LDS ----
    for (int e = t; e < EPER; e += 256) {
        unsigned pk = epack[e];
        atomicAdd(&scacc[pk >> 16], srel[pk & 0xFFFFu]);
    }
    __syncthreads();

    const float brel = sag_brel[0];
    for (int node = t; node < NN; node += 256)
        scoreS[node] = tanhf(scacc[node] + brel + sroot[node]);
    __syncthreads();

    // ---- phase 7: top-K=200 of 400 by exact rank (ties -> lower index) ----
    for (int i = t; i < NN; i += 256) {
        float si = scoreS[i];
        int rank = 0;
        for (int j = 0; j < NN; j++) {
            float sj = scoreS[j];
            rank += (sj > si) || (sj == si && j < i);
        }
        wsel[i] = (rank < KSEL) ? si : 0.0f;   // unselected contribute 0
    }
    __syncthreads();

    // ---- phase 8: emb = mean over selected of h3 * score ----
    {
        int f = t & 63, chunk = t >> 6;
        float acc = 0.0f;
        for (int node = chunk * 100; node < (chunk + 1) * 100; node++)
            acc += wsel[node] * h3[node * HH + f];
        part[chunk * HH + f] = acc;
    }
    __syncthreads();
    if (t < HH) {
        float s = part[t] + part[HH + t] + part[2 * HH + t] + part[3 * HH + t];
        emb_out[(size_t)p * HH + t] = s * (1.0f / KSEL);
    }
}

// ---------------- Kernel B: PPI pair decoder ----------------
// dynamic LDS layout (bytes):
#define B_OFF_PAIR 0        // [64][128] f32 = 32768
#define B_OFF_FEAT 32768    // [64][256] f16 = 32768
#define B_OFF_W16  65536    // [256][64] f16 = 32768
#define B_OFF_Z    98304    // [64][64]  f32 = 16384
#define SMEM_B     114688

__global__ __launch_bounds__(256) void ppi_decoder_kernel(
    const float* __restrict__ emb, const int* __restrict__ ppi_idx,
    const float* __restrict__ conv_w, const float* __restrict__ conv_b,
    const float* __restrict__ lin1_w, const float* __restrict__ lin1_b,
    const float* __restrict__ lin2_w, const float* __restrict__ lin2_b,
    float* __restrict__ ppi_out)
{
    extern __shared__ char smem[];
    float*    pairb = (float*)(smem + B_OFF_PAIR);     // [64][128]
    _Float16* feat  = (_Float16*)(smem + B_OFF_FEAT);  // [64][256]
    _Float16* w16   = (_Float16*)(smem + B_OFF_W16);   // [256][64]
    float*    zb    = (float*)(smem + B_OFF_Z);        // [64][64]

    __shared__ float cw[12], cb[4], lb[HH], w2[HH];
    __shared__ float red[256];

    const int t = threadIdx.x;
    const int m0 = blockIdx.x * 64;
    const int lane = t & 31;
    const int wave = __builtin_amdgcn_readfirstlane(t >> 5);
    const int mrow = lane & 15, khalf = lane >> 4;

    // ---- async DMA gather: pair = concat(emb[a], emb[b]) (16B/lane chunks) ----
    // 64 rows x 2 halves x 16 chunks = 2048 chunks; overlapped with weight conversion
    {
        unsigned lbase = lds_off_of(pairb);
        for (int c = t; c < 2048; c += 256) {
            int row   = c >> 5;          // 32 chunks per row
            int half  = (c >> 4) & 1;    // 0: emb[a], 1: emb[b]
            int chunk = c & 15;          // 16B units within a 256B embedding row
            int prot  = ppi_idx[(half ? MM : 0) + m0 + row];
            unsigned long long ga = (unsigned long long)(const void*)(emb + (size_t)prot * HH)
                                    + 16ull * chunk;
            async_copy_b128(lbase + (unsigned)(row * 512 + half * 256 + chunk * 16), ga);
        }
    }

    if (t < 12) cw[t] = conv_w[t];
    if (t < 4)  cb[t] = conv_b[t];
    if (t >= 32 && t < 32 + HH) { lb[t - 32] = lin1_b[t - 32]; w2[t - 32] = lin2_w[t - 32]; }
    for (int i = t; i < 256 * HH; i += 256) w16[i] = (_Float16)lin1_w[i];

    wait_asynccnt0();
    __syncthreads();

    // conv1d(k=3,pad=1,4ch) + maxpool(2) -> feat [64][4*64] f16
    for (int i = t; i < 64 * 4 * 64; i += 256) {
        int r = i >> 8, ch = (i >> 6) & 3, j = i & 63;
        float o0 = cb[ch], o1 = cb[ch];
#pragma unroll
        for (int tt = 0; tt < 3; tt++) {
            int i0 = 2 * j - 1 + tt;
            int i1 = 2 * j + tt;
            float wv = cw[ch * 3 + tt];
            if (i0 >= 0 && i0 < 128) o0 += pairb[r * 128 + i0] * wv;
            if (i1 < 128)            o1 += pairb[r * 128 + i1] * wv;
        }
        feat[r * 256 + ch * 64 + j] = (_Float16)fmaxf(o0, o1);
    }
    __syncthreads();

    // lin1: [64,256]x[256,64] -> 16 tiles of 16x16, K=256 (8 WMMAs/tile)
    for (int tile = wave; tile < 16; tile += 8) {
        int mt = (tile >> 2) * 16;
        int nt = (tile & 3) * 16;
        v8f c = {};
#pragma unroll
        for (int kb = 0; kb < 256; kb += 32) {
            v16h a, b;
#pragma unroll
            for (int e = 0; e < 16; e++) {
                a[e] = feat[(mt + mrow) * 256 + kb + amap(e, khalf)];
                b[e] = w16[(kb + bmap(e, khalf)) * HH + nt + mrow];
            }
            c = __builtin_amdgcn_wmma_f32_16x16x32_f16(false, a, false, b,
                                                       (short)0, c, false, false);
        }
        int f = nt + mrow;
        float bv = lb[f];
#pragma unroll
        for (int r = 0; r < 8; r++) {
            int m = mt + r + 8 * khalf;
            zb[m * HH + f] = fmaxf(c[r] + bv, 0.0f);
        }
    }
    __syncthreads();

    // lin2 + sigmoid: 4 threads per row
    {
        int row = t >> 2, q = t & 3;
        float acc = 0.0f;
#pragma unroll
        for (int k = q * 16; k < q * 16 + 16; k++) acc += zb[row * HH + k] * w2[k];
        red[t] = acc;
    }
    __syncthreads();
    if ((t & 3) == 0) {
        int row = t >> 2;
        float s = red[t] + red[t + 1] + red[t + 2] + red[t + 3] + lin2_b[0];
        ppi_out[m0 + row] = 1.0f / (1.0f + expf(-s));
    }
}

// ---------------- launcher ----------------
extern "C" void kernel_launch(void* const* d_in, const int* in_sizes, int n_in,
                              void* d_out, int out_size, void* d_ws, size_t ws_size,
                              hipStream_t stream) {
    const float* x        = (const float*)d_in[0];
    const int*   ei       = (const int*)d_in[1];
    const int*   ppi_idx  = (const int*)d_in[2];
    const float* sage_wl  = (const float*)d_in[3];
    const float* sage_wr  = (const float*)d_in[4];
    const float* sage_b   = (const float*)d_in[5];
    const float* fc1_w    = (const float*)d_in[6];
    const float* fc1_b    = (const float*)d_in[7];
    const float* bn_g     = (const float*)d_in[8];
    const float* bn_b     = (const float*)d_in[9];
    const float* bn_rm    = (const float*)d_in[10];
    const float* bn_rv    = (const float*)d_in[11];
    const float* sag_wrel = (const float*)d_in[12];
    const float* sag_brel = (const float*)d_in[13];
    const float* sag_wroot= (const float*)d_in[14];
    const float* conv_w   = (const float*)d_in[15];
    const float* conv_b   = (const float*)d_in[16];
    const float* lin1_w   = (const float*)d_in[17];
    const float* lin1_b   = (const float*)d_in[18];
    const float* lin2_w   = (const float*)d_in[19];
    const float* lin2_b   = (const float*)d_in[20];

    float* emb_out = (float*)d_out;                   // [1512*64]
    float* ppi_out = (float*)d_out + (size_t)PP * HH; // [65536]

    protein_gnn_kernel<<<PP, 256, SMEM_A, stream>>>(
        x, ei, sage_wl, sage_wr, sage_b, fc1_w, fc1_b,
        bn_g, bn_b, bn_rm, bn_rv, sag_wrel, sag_brel, sag_wroot, emb_out);

    ppi_decoder_kernel<<<MM / 64, 256, SMEM_B, stream>>>(
        emb_out, ppi_idx, conv_w, conv_b, lin1_w, lin1_b, lin2_w, lin2_b, ppi_out);
}
